// EdgeRolandGNN_20418274525539
// MI455X (gfx1250) — matrix operand order repaired and checked by reading, not compiled, not run on hardware
//
#include <hip/hip_runtime.h>
#include <hip/hip_bf16.h>

// ---------------------------------------------------------------------------
// Types for CDNA5 WMMA (wave32): v_wmma_f32_16x16x32_bf16
// ---------------------------------------------------------------------------
typedef __attribute__((ext_vector_type(16))) __bf16 v16bf;
typedef __attribute__((ext_vector_type(8)))  __bf16 v8bf;
typedef __attribute__((ext_vector_type(8)))  float  v8f;

#if defined(__gfx1250__) && __has_builtin(__builtin_amdgcn_global_load_async_to_lds_b128) && __has_builtin(__builtin_amdgcn_s_wait_asynccnt)
#define USE_ASYNC_LDS 1
#else
#define USE_ASYNC_LDS 0
#endif

__device__ __forceinline__ __bf16 f2bf(float f) {
    unsigned u = __builtin_bit_cast(unsigned, f);
    unsigned r = u + 0x7FFFu + ((u >> 16) & 1u);   // round-to-nearest-even
    unsigned short h = (unsigned short)(r >> 16);
    return __builtin_bit_cast(__bf16, h);
}

__device__ __forceinline__ float leaky01(float v) {
    return v >= 0.0f ? v : 0.01f * v;
}

// ---------------------------------------------------------------------------
// Generic fill-zero / f32 -> bf16 convert
// ---------------------------------------------------------------------------
__global__ void zero_kernel(float* __restrict__ p, long n) {
    long i = (long)blockIdx.x * blockDim.x + threadIdx.x;
    if (i < n) p[i] = 0.0f;
}

__global__ void cvt_bf16_kernel(const float* __restrict__ in, __bf16* __restrict__ out, long n) {
    long i = (long)blockIdx.x * blockDim.x + threadIdx.x;
    if (i < n) out[i] = f2bf(in[i]);
}

// ---------------------------------------------------------------------------
// Pack B [K,N] f32 row-major -> bf16 WMMA fragment layout:
//   Bp[ ((tileN*(K/32) + kt)*32 + lane)*16 + e ],
//   lane = (hi<<4)|(n&15), k = kt*32 + hi*16 + e
// Each lane's fragment is then one contiguous 32B chunk.
// ---------------------------------------------------------------------------
__global__ void pack_B_kernel(const float* __restrict__ B, __bf16* __restrict__ Bp,
                              int K, int N) {
    int idx = blockIdx.x * blockDim.x + threadIdx.x;
    if (idx >= K * N) return;
    int k = idx / N, n = idx - k * N;
    int kt = k >> 5, kk = k & 31, hi = kk >> 4, e = kk & 15;
    int tn = n >> 4, L = (hi << 4) | (n & 15);
    long p = (((long)tn * (K >> 5) + kt) * 32 + L) * 16 + e;
    Bp[p] = f2bf(B[idx]);
}

// ---------------------------------------------------------------------------
// Degree: atomic count of dst occurrences, then dis = rsqrt(deg + 1)
// ---------------------------------------------------------------------------
__global__ void deg_count_kernel(const int* __restrict__ dst, float* __restrict__ deg, int E) {
    int e = blockIdx.x * blockDim.x + threadIdx.x;
    if (e < E) atomicAdd(&deg[dst[e]], 1.0f);
}

__global__ void deg_rsqrt_kernel(float* __restrict__ dis, int n) {
    int i = blockIdx.x * blockDim.x + threadIdx.x;
    if (i < n) dis[i] = rsqrtf(dis[i] + 1.0f);
}

// ---------------------------------------------------------------------------
// WMMA GEMM, bf16 inputs: C[M,Ncols] = act(A[M,K] * B + bias)
//   - A: bf16 row-major, 16B-aligned rows (K multiple of 32)
//   - Bp: pre-packed fragment layout (see pack_B_kernel)
//   - one wave -> 16x32 output (two 16x16 tiles, shared A fragment)
//   - B slab for the block's 32 columns staged in LDS (async if available)
//   - K fully unrolled: K/32 * 2 wmma per wave
// ---------------------------------------------------------------------------
template <int K, bool OUTBF>
__global__ void wmma_gemm_bf16(const __bf16* __restrict__ A,
                               const __bf16* __restrict__ Bp,
                               const float* __restrict__ bias,   // may be null
                               void* __restrict__ Cout,
                               int M, int Ncols, int do_leaky) {
    constexpr int KT = K / 32;
    __shared__ __align__(32) __bf16 ldsB[2 * KT * 512];   // two 16-col tiles

    // ---- stage packed B slab (columns [32*blockIdx.y, +32)) into LDS ----
    {
        const uint4* gsrc = (const uint4*)(Bp + (long)blockIdx.y * (2 * KT * 512));
        uint4* ldst = (uint4*)ldsB;
        const int chunks = KT * 128;                       // (2*KT*512*2B)/16B
#if USE_ASYNC_LDS
        typedef int async_v4i __attribute__((__vector_size__(4 * sizeof(int))));
        typedef __attribute__((address_space(1))) async_v4i g_v4i;
        typedef __attribute__((address_space(3))) async_v4i l_v4i;
        for (int i = threadIdx.x; i < chunks; i += blockDim.x)
            __builtin_amdgcn_global_load_async_to_lds_b128(
                (g_v4i*)(gsrc + i), (l_v4i*)(ldst + i), 0, 0);
        __builtin_amdgcn_s_wait_asynccnt(0);
#else
        for (int i = threadIdx.x; i < chunks; i += blockDim.x)
            ldst[i] = gsrc[i];
#endif
    }
    __syncthreads();

    const int wave = threadIdx.x >> 5;                     // 8 waves / block
    const int lane = threadIdx.x & 31;
    const int tileM = blockIdx.x * 8 + wave;
    if (tileM * 16 >= M) return;
    const int m  = lane & 15;
    const int hi = lane >> 4;

    const __bf16* arow = A + (long)(tileM * 16 + m) * K;
    v8f acc0 = {}, acc1 = {};

    #pragma unroll
    for (int kt = 0; kt < KT; ++kt) {
        // A 16x32 bf16 fragment: two contiguous 16B chunks per lane
        v8bf alo = *(const v8bf*)(arow + kt * 32 + hi * 16);
        v8bf ahi = *(const v8bf*)(arow + kt * 32 + hi * 16 + 8);
        v16bf afrag;
        #pragma unroll
        for (int i = 0; i < 8; ++i) { afrag[i] = alo[i]; afrag[i + 8] = ahi[i]; }

        // B fragments: contiguous 32B per lane from LDS
        v16bf bf0 = *(const v16bf*)(ldsB + ((0 * KT + kt) * 32 + lane) * 16);
        v16bf bf1 = *(const v16bf*)(ldsB + ((1 * KT + kt) * 32 + lane) * 16);

        acc0 = __builtin_amdgcn_wmma_f32_16x16x32_bf16(
                   false, afrag, false, bf0, (short)0, acc0, false, false);
        acc1 = __builtin_amdgcn_wmma_f32_16x16x32_bf16(
                   false, afrag, false, bf1, (short)0, acc1, false, false);
    }

    // C/D f32 16x16: vgpr r -> row M = r + 8*hi, col = lane&15
    const int n0 = blockIdx.y * 32 + (lane & 15);
    const int n1 = n0 + 16;
    const float bv0 = bias ? bias[n0] : 0.0f;
    const float bv1 = bias ? bias[n1] : 0.0f;
    #pragma unroll
    for (int r = 0; r < 8; ++r) {
        const long mm = (long)(tileM * 16 + r + hi * 8);
        float v0 = acc0[r] + bv0;
        float v1 = acc1[r] + bv1;
        if (do_leaky) { v0 = leaky01(v0); v1 = leaky01(v1); }
        if (OUTBF) {
            ((__bf16*)Cout)[mm * Ncols + n0] = f2bf(v0);
            ((__bf16*)Cout)[mm * Ncols + n1] = f2bf(v1);
        } else {
            ((float*)Cout)[mm * Ncols + n0] = v0;
            ((float*)Cout)[mm * Ncols + n1] = v1;
        }
    }
}

// ---------------------------------------------------------------------------
// Edge scatter: agg[dst] += xw[src] * dis[src]*dis[dst]   (C = 128 or 64)
// ---------------------------------------------------------------------------
__global__ void scatter_kernel(const int* __restrict__ src, const int* __restrict__ dst,
                               const float* __restrict__ dis,
                               const float* __restrict__ xw, float* __restrict__ agg,
                               int E, int C) {
    const int perEdge     = C >> 2;                    // float4 chunks per edge
    const int edgesPerBlk = blockDim.x / perEdge;
    const int eLocal = threadIdx.x / perEdge;
    const int chunk  = threadIdx.x - eLocal * perEdge;
    const long e = (long)blockIdx.x * edgesPerBlk + eLocal;
    if (e >= E) return;
    const int s = src[e], d = dst[e];
    const float coeff = dis[s] * dis[d];
    const float4 x4 = ((const float4*)(xw + (long)s * C))[chunk];
    float* out = agg + (long)d * C + chunk * 4;
    atomicAdd(out + 0, x4.x * coeff);
    atomicAdd(out + 1, x4.y * coeff);
    atomicAdd(out + 2, x4.z * coeff);
    atomicAdd(out + 3, x4.w * coeff);
}

// ---------------------------------------------------------------------------
// Finalize GCN layer: emb = tau*prev + (1-tau)*leaky(agg + xw*dis^2 + b)
// ---------------------------------------------------------------------------
__global__ void finalize_kernel(const float* __restrict__ agg, const float* __restrict__ xw,
                                const float* __restrict__ dis, const float* __restrict__ bias,
                                const float* __restrict__ prev, float* __restrict__ emb,
                                const int* __restrict__ ncur, const int* __restrict__ nprev,
                                long total, int C) {
    long gid = (long)blockIdx.x * blockDim.x + threadIdx.x;
    if (gid >= total) return;
    const long node = gid / C;
    const int  c    = (int)(gid - node * C);
    const float np = (float)(*nprev), nc = (float)(*ncur);
    const float tau = np / (np + nc);
    const float di = dis[node];
    float v = agg[gid] + xw[gid] * di * di + bias[c];
    v = leaky01(v);
    emb[gid] = tau * prev[gid] + (1.0f - tau) * v;
}

// ---------------------------------------------------------------------------
// Edge scoring: score[q] = sum_c emb2[a][c]*emb2[b][c]*(Wp[c][0]+Wp[c][1]) + bp0+bp1
// ---------------------------------------------------------------------------
__global__ void score_kernel(const float* __restrict__ emb2,
                             const int* __restrict__ qsrc, const int* __restrict__ qdst,
                             const float* __restrict__ W_post, const float* __restrict__ b_post,
                             float* __restrict__ out, int EQ, int C) {
    int q = blockIdx.x * blockDim.x + threadIdx.x;
    if (q >= EQ) return;
    const float* ha = emb2 + (long)qsrc[q] * C;
    const float* hb = emb2 + (long)qdst[q] * C;
    float s = b_post[0] + b_post[1];
    for (int c = 0; c < C; ++c)
        s += ha[c] * hb[c] * (W_post[2 * c] + W_post[2 * c + 1]);
    out[q] = s;
}

// ---------------------------------------------------------------------------
// Launch
// ---------------------------------------------------------------------------
extern "C" void kernel_launch(void* const* d_in, const int* in_sizes, int n_in,
                              void* d_out, int out_size, void* d_ws, size_t ws_size,
                              hipStream_t stream) {
    const float* x      = (const float*)d_in[0];
    const int*   ei     = (const int*)d_in[1];     // [2, E]
    const int*   eli    = (const int*)d_in[2];     // [2, EQ]
    const float* prev1  = (const float*)d_in[3];
    const float* prev2  = (const float*)d_in[4];
    const float* W_pre1 = (const float*)d_in[5];
    const float* b_pre1 = (const float*)d_in[6];
    const float* W_pre2 = (const float*)d_in[7];
    const float* b_pre2 = (const float*)d_in[8];
    const float* W_c1   = (const float*)d_in[9];
    const float* b_c1   = (const float*)d_in[10];
    const float* W_c2   = (const float*)d_in[11];
    const float* b_c2   = (const float*)d_in[12];
    const float* W_post = (const float*)d_in[13];
    const float* b_post = (const float*)d_in[14];
    const int*   ncur   = (const int*)d_in[15];
    const int*   nprev  = (const int*)d_in[16];

    const int D_IN = 128, HPRE = 256, H1 = 128, H2 = 64;
    const int Nn = in_sizes[0] / D_IN;          // 50000
    const int E  = in_sizes[1] / 2;             // 600000
    const int EQ = in_sizes[2] / 2;             // 200000
    const int* src  = ei;
    const int* dst  = ei + E;
    const int* qsrc = eli;
    const int* qdst = eli + EQ;

    // ---- workspace layout (region reuse) ----
    char* w = (char*)d_ws;
    float* dis = (float*)w;                 w += (size_t)Nn * 4;           // deg->dis
    __bf16* RA = (__bf16*)w;                w += (size_t)Nn * 128 * 2;     // xbf, later emb1bf
    __bf16* RB = (__bf16*)w;                w += (size_t)Nn * 256 * 2;     // h1bf
    __bf16* RC = (__bf16*)w;                w += (size_t)Nn * 128 * 2;     // h2bf
    float*  F  = (float*)w;                 w += (size_t)Nn * 256 * 4;     // xw|agg regions
    __bf16* Bp1 = (__bf16*)w;               w += (size_t)D_IN * HPRE * 2;  // packed W_pre1
    __bf16* Bp2 = (__bf16*)w;               w += (size_t)HPRE * H1 * 2;    // packed W_pre2
    __bf16* Bp3 = (__bf16*)w;               w += (size_t)H1 * H1 * 2;      // packed W_c1
    __bf16* Bp4 = (__bf16*)w;               w += (size_t)H1 * H2 * 2;      // packed W_c2

    __bf16* xbf    = RA;
    __bf16* emb1bf = RA;        // reuses xbf region (xbf dead after MLP1)
    __bf16* h1bf   = RB;
    __bf16* h2bf   = RC;
    float* xw1  = F;
    float* agg1 = F + (size_t)Nn * H1;
    float* xw2  = F;            // reuse after finalize1
    float* agg2 = F + (size_t)Nn * H2;

    // Output layout: scores[EQ] | emb1[Nn*H1] | emb2[Nn*H2]
    float* scores = (float*)d_out;
    float* emb1   = scores + EQ;
    float* emb2   = emb1 + (size_t)Nn * H1;

    const int tilesM = Nn / 16;                 // 3125 (exact)
    const dim3 gblk((tilesM + 7) / 8, 1, 1);

    // ---- pre-pack weights + convert x (independent) ----
    pack_B_kernel<<<(D_IN * HPRE + 255) / 256, 256, 0, stream>>>(W_pre1, Bp1, D_IN, HPRE);
    pack_B_kernel<<<(HPRE * H1 + 255) / 256, 256, 0, stream>>>(W_pre2, Bp2, HPRE, H1);
    pack_B_kernel<<<(H1 * H1 + 255) / 256, 256, 0, stream>>>(W_c1, Bp3, H1, H1);
    pack_B_kernel<<<(H1 * H2 + 255) / 256, 256, 0, stream>>>(W_c2, Bp4, H1, H2);
    cvt_bf16_kernel<<<(int)(((long)Nn * D_IN + 255) / 256), 256, 0, stream>>>(x, xbf, (long)Nn * D_IN);

    // ---- degree / normalization ----
    zero_kernel<<<(Nn + 255) / 256, 256, 0, stream>>>(dis, (long)Nn);
    deg_count_kernel<<<(E + 255) / 256, 256, 0, stream>>>(dst, dis, E);
    deg_rsqrt_kernel<<<(Nn + 255) / 256, 256, 0, stream>>>(dis, Nn);

    // ---- preprocess MLP (WMMA, bf16 chained activations) ----
    wmma_gemm_bf16<128, true><<<dim3(gblk.x, HPRE / 32), 256, 0, stream>>>(
        xbf, Bp1, b_pre1, h1bf, Nn, HPRE, 1);
    wmma_gemm_bf16<256, true><<<dim3(gblk.x, H1 / 32), 256, 0, stream>>>(
        h1bf, Bp2, b_pre2, h2bf, Nn, H1, 1);

    // ---- GCN layer 1 ----
    wmma_gemm_bf16<128, false><<<dim3(gblk.x, H1 / 32), 256, 0, stream>>>(
        h2bf, Bp3, nullptr, xw1, Nn, H1, 0);
    zero_kernel<<<(int)(((long)Nn * H1 + 255) / 256), 256, 0, stream>>>(agg1, (long)Nn * H1);
    {
        int perEdge = H1 / 4, edgesPerBlk = 256 / perEdge;
        int blocks = (E + edgesPerBlk - 1) / edgesPerBlk;
        scatter_kernel<<<blocks, 256, 0, stream>>>(src, dst, dis, xw1, agg1, E, H1);
    }
    finalize_kernel<<<(int)(((long)Nn * H1 + 255) / 256), 256, 0, stream>>>(
        agg1, xw1, dis, b_c1, prev1, emb1, ncur, nprev, (long)Nn * H1, H1);
    cvt_bf16_kernel<<<(int)(((long)Nn * H1 + 255) / 256), 256, 0, stream>>>(
        emb1, emb1bf, (long)Nn * H1);

    // ---- GCN layer 2 ----
    wmma_gemm_bf16<128, false><<<dim3(gblk.x, H2 / 32), 256, 0, stream>>>(
        emb1bf, Bp4, nullptr, xw2, Nn, H2, 0);
    zero_kernel<<<(int)(((long)Nn * H2 + 255) / 256), 256, 0, stream>>>(agg2, (long)Nn * H2);
    {
        int perEdge = H2 / 4, edgesPerBlk = 256 / perEdge;
        int blocks = (E + edgesPerBlk - 1) / edgesPerBlk;
        scatter_kernel<<<blocks, 256, 0, stream>>>(src, dst, dis, xw2, agg2, E, H2);
    }
    finalize_kernel<<<(int)(((long)Nn * H2 + 255) / 256), 256, 0, stream>>>(
        agg2, xw2, dis, b_c2, prev2, emb2, ncur, nprev, (long)Nn * H2, H2);

    // ---- edge scoring ----
    score_kernel<<<(EQ + 255) / 256, 256, 0, stream>>>(emb2, qsrc, qdst, W_post, b_post,
                                                       scores, EQ, H2);
}
